// AnchorDiffNet_86363202388630
// MI455X (gfx1250) — matrix operand order repaired and checked by reading, not compiled, not run on hardware
//
#include <hip/hip_runtime.h>

#define BATCH 8
#define CHAN 128
#define MPIX 4096          // 64*64 pixels per batch
#define TILE_N 64          // key tile
#define NTILES (MPIX / TILE_N)
#define LDA_KV_CN 72       // [c][key] stride (bf16 elems): 144B rows, 16B-aligned frags
#define LDA_KV_NC 136      // [key][c] stride: 272B rows
#define LDA_P 72           // per-wave P strip stride
#define LDA_X 392          // xT row stride (384 channels + pad): 784B rows

typedef __attribute__((ext_vector_type(16))) __bf16 v16bf;
typedef __attribute__((ext_vector_type(8)))  float  v8f;
typedef __attribute__((ext_vector_type(4)))  int    v4i;   // native vector for builtins/LDS frags

union Frag16 { v16bf bf; unsigned short u[16]; v4i q[2]; };

#define AS1 __attribute__((address_space(1)))
#define AS3 __attribute__((address_space(3)))

#if defined(__has_builtin)
#if __has_builtin(__builtin_amdgcn_global_load_async_to_lds_b128) && \
    __has_builtin(__builtin_amdgcn_s_wait_asynccnt)
#define HAVE_ASYNC_LDS 1
#endif
#endif
#ifndef HAVE_ASYNC_LDS
#define HAVE_ASYNC_LDS 0
#endif

__device__ __forceinline__ unsigned short f2bf(float x) {
  unsigned int u = __float_as_uint(x);
  u += 0x7FFFu + ((u >> 16) & 1u);           // round-to-nearest-even
  return (unsigned short)(u >> 16);
}
__device__ __forceinline__ float bf2f(unsigned short h) {
  return __uint_as_float(((unsigned int)h) << 16);
}

// 16-byte global(bf16) -> LDS copy; async (ASYNCcnt) on CDNA5.
__device__ __forceinline__ void copy16(const unsigned short* g, unsigned short* l) {
#if HAVE_ASYNC_LDS
  __builtin_amdgcn_global_load_async_to_lds_b128((AS1 v4i*)g, (AS3 v4i*)l, 0, 0);
#else
  *(v4i*)l = *(const v4i*)g;
#endif
}
__device__ __forceinline__ void wait_async() {
#if HAVE_ASYNC_LDS
  __builtin_amdgcn_s_wait_asynccnt(0);
#endif
}

// ---- dynamic LDS layout (ushort units) ----
constexpr int KV_CN_WORDS = CHAN * LDA_KV_CN;            // 9216
constexpr int KV_NC_WORDS = TILE_N * LDA_KV_NC;          // 8704
constexpr int P_WORDS     = 16 * LDA_P;                  // 1152
constexpr int OFF_KV_NC   = 2 * KV_CN_WORDS;             // double-buffered
constexpr int OFF_P       = OFF_KV_NC + 2 * KV_NC_WORDS;
constexpr int OFF_RED     = OFF_P + 8 * P_WORDS;         // 45056 words
constexpr size_t SHM_BYTES = (size_t)OFF_RED * 2 + 256 * 4;   // ~91 KB (320KB/WGP)

// stage one 128x64 K/V tile (both orientations) into LDS buffers
template <int PRE>
__device__ __forceinline__ void stage_tile(unsigned short* kvcn, unsigned short* kvnc,
                                           const unsigned short* gcm, const unsigned short* gmc,
                                           const float* curb, int n0, int tid) {
  if (PRE) {
    for (int id = tid; id < 1024; id += 256) {           // [c][key] rows: 8 keys / chunk
      int c = id >> 3, k8 = id & 7;
      copy16(gcm + (size_t)c * MPIX + n0 + k8 * 8, kvcn + c * LDA_KV_CN + k8 * 8);
    }
    for (int id = tid; id < 1024; id += 256) {           // [key][c] rows: 8 chans / chunk
      int key = id >> 4, c8 = id & 15;
      copy16(gmc + (size_t)(n0 + key) * CHAN + c8 * 8, kvnc + key * LDA_KV_NC + c8 * 8);
    }
  } else {
    for (int idx = tid; idx < CHAN * TILE_N; idx += 256) {
      int c = idx >> 6, k = idx & 63;
      unsigned short v = f2bf(curb[(size_t)c * MPIX + n0 + k]);
      kvcn[c * LDA_KV_CN + k] = v;
      kvnc[k * LDA_KV_NC + c] = v;
    }
  }
}

template <int PRE>
__global__ __launch_bounds__(256)
void anchor_attn_head(const float* __restrict__ refp,
                      const float* __restrict__ curp,
                      const float* __restrict__ w1,
                      const float* __restrict__ b1,
                      const float* __restrict__ gamma,
                      const float* __restrict__ beta,
                      const float* __restrict__ rmean,
                      const float* __restrict__ rvar,
                      const float* __restrict__ w2,
                      const float* __restrict__ b2,
                      float* __restrict__ out,
                      const unsigned short* __restrict__ curCM,   // bf16 [b][c][m]
                      const unsigned short* __restrict__ curMC,   // bf16 [b][m][c]
                      const unsigned short* __restrict__ refMC,   // bf16 [b][m][c]
                      const unsigned short* __restrict__ w1bf)    // bf16 [o][384]
{
  extern __shared__ unsigned short smem[];
  unsigned short* Pbuf = smem + OFF_P;
  float* red = (float*)(smem + OFF_RED);
  unsigned short* xT = smem;                 // phase-2 overlay: [pix][384] bf16

  const int tid  = threadIdx.x;
  const int lane = tid & 31;
  const int wave = tid >> 5;
  const int ln   = lane & 15;
  const int hi   = lane >> 4;
  const int a    = wave >> 2;   // 0: cross (Q=ref), 1: self (Q=cur)
  const int wq   = wave & 3;    // 16-row query sub-tile

  const int b  = blockIdx.y;
  const int m0 = blockIdx.x * 64;

  const float* curb = curp + (size_t)b * CHAN * MPIX;
  const unsigned short* gcm = PRE ? curCM + (size_t)b * CHAN * MPIX : nullptr;
  const unsigned short* gmc = PRE ? curMC + (size_t)b * CHAN * MPIX : nullptr;

  const int gm = m0 + wq * 16 + ln;          // this lane's query pixel (A-layout row)

  // ---- Q fragments: 16x32 bf16 A-layout, 4 chunks over the 128 channels ----
  Frag16 qf[4];
  if (PRE) {
    const unsigned short* qmc = (a == 0 ? refMC : curMC) + (size_t)b * CHAN * MPIX;
#pragma unroll
    for (int kc = 0; kc < 4; ++kc) {
      const unsigned short* p = qmc + (size_t)gm * CHAN + kc * 32 + hi * 8;
      qf[kc].q[0] = *(const v4i*)p;
      qf[kc].q[1] = *(const v4i*)(p + 16);
    }
  } else {
    const float* qsrc = (a == 0 ? refp : curp) + (size_t)b * CHAN * MPIX;
#pragma unroll
    for (int kc = 0; kc < 4; ++kc)
#pragma unroll
      for (int i = 0; i < 16; ++i) {
        int c = kc * 32 + (i & 7) + ((i >> 3) << 4) + hi * 8;
        qf[kc].u[i] = f2bf(qsrc[(size_t)c * MPIX + gm]);
      }
  }

  v8f zero = {};
  v8f Oacc[8];
#pragma unroll
  for (int ct = 0; ct < 8; ++ct) Oacc[ct] = zero;
  float rowm[8], rowl[8];                    // running max kept in RAW (unscaled) domain
#pragma unroll
  for (int r = 0; r < 8; ++r) { rowm[r] = -__builtin_inff(); rowl[r] = 0.f; }

  const float scale2 = 0.12751738f;          // 128^-0.5 * log2(e): softmax in base-2

  // prefetch tile 0
  stage_tile<PRE>(smem, smem + OFF_KV_NC, gcm, gmc, curb, 0, tid);

  for (int it = 0; it < NTILES; ++it) {
    wait_async();
    __syncthreads();                         // tile `it` ready; tile `it-1` compute done
    const int p = it & 1;
    unsigned short* kv_cn = smem + p * KV_CN_WORDS;
    unsigned short* kv_nc = smem + OFF_KV_NC + p * KV_NC_WORDS;
    if (it + 1 < NTILES)                     // overlap next-tile copy with compute
      stage_tile<PRE>(smem + (1 - p) * KV_CN_WORDS,
                      smem + OFF_KV_NC + (1 - p) * KV_NC_WORDS,
                      gcm, gmc, curb, (it + 1) * TILE_N, tid);

    // ---- S = Q K^T : 16x64 slice per wave (B-frag loads pipelined 1 ahead) ----
    v8f st[4];
#pragma unroll
    for (int t = 0; t < 4; ++t) {
      const int base0 = (t * 16 + ln) * LDA_KV_NC + hi * 8;
      Frag16 bb[4];
      bb[0].q[0] = *(const v4i*)&kv_nc[base0];
      bb[0].q[1] = *(const v4i*)&kv_nc[base0 + 16];
      v8f acc = zero;
#pragma unroll
      for (int kc = 0; kc < 4; ++kc) {
        if (kc < 3) {
          int bbase = base0 + (kc + 1) * 32;
          bb[kc + 1].q[0] = *(const v4i*)&kv_nc[bbase];
          bb[kc + 1].q[1] = *(const v4i*)&kv_nc[bbase + 16];
        }
        acc = __builtin_amdgcn_wmma_f32_16x16x32_bf16(false, qf[kc].bf, false, bb[kc].bf,
                                                      (short)0, acc, false, false);
      }
      st[t] = acc;                           // raw scores; scale folded into exponent
    }

    // ---- online softmax (base-2), shfl_xor within 16-lane halves ----
#pragma unroll
    for (int r = 0; r < 8; ++r) {
      float v = fmaxf(fmaxf(st[0][r], st[1][r]), fmaxf(st[2][r], st[3][r]));
#pragma unroll
      for (int mk = 1; mk < 16; mk <<= 1) v = fmaxf(v, __shfl_xor(v, mk, 32));
      float mnew = fmaxf(rowm[r], v);
      float msc  = mnew * scale2;
      float corr = exp2f(fmaf(rowm[r], scale2, -msc));
      rowm[r] = mnew;
      float ps = 0.f;
#pragma unroll
      for (int t = 0; t < 4; ++t) {
        float pv = exp2f(fmaf(st[t][r], scale2, -msc));
        st[t][r] = pv;
        ps += pv;
      }
#pragma unroll
      for (int mk = 1; mk < 16; mk <<= 1) ps += __shfl_xor(ps, mk, 32);
      rowl[r] = rowl[r] * corr + ps;
#pragma unroll
      for (int ct = 0; ct < 8; ++ct) Oacc[ct][r] *= corr;
    }

    // ---- stage P bf16 row-major [m][key] (per-wave strip) ----
    unsigned short* Pw = Pbuf + wave * P_WORDS;
#pragma unroll
    for (int t = 0; t < 4; ++t)
#pragma unroll
      for (int r = 0; r < 8; ++r)
        Pw[(r + 8 * hi) * LDA_P + t * 16 + ln] = f2bf(st[t][r]);

    // ---- O += P V  (V-frag loads pipelined 1 ahead) ----
#pragma unroll
    for (int kt = 0; kt < 2; ++kt) {
      Frag16 pa;
      int abase = ln * LDA_P + kt * 32 + hi * 8;
      pa.q[0] = *(const v4i*)&Pw[abase];
      pa.q[1] = *(const v4i*)&Pw[abase + 16];
      Frag16 vb[8];
      const int vbase0 = ln * LDA_KV_CN + kt * 32 + hi * 8;
      vb[0].q[0] = *(const v4i*)&kv_cn[vbase0];
      vb[0].q[1] = *(const v4i*)&kv_cn[vbase0 + 16];
#pragma unroll
      for (int ct = 0; ct < 8; ++ct) {
        if (ct < 7) {
          int bbase = vbase0 + (ct + 1) * 16 * LDA_KV_CN;
          vb[ct + 1].q[0] = *(const v4i*)&kv_cn[bbase];
          vb[ct + 1].q[1] = *(const v4i*)&kv_cn[bbase + 16];
        }
        Oacc[ct] = __builtin_amdgcn_wmma_f32_16x16x32_bf16(false, pa.bf, false, vb[ct].bf,
                                                           (short)0, Oacc[ct], false, false);
      }
    }
  }

  __syncthreads();                           // key loop done; reuse LDS as xT

  // ---- xT[pix][0:128]=feats0, [128:256]=feats1, [256:384]=cur (bf16) ----
#pragma unroll
  for (int ct = 0; ct < 8; ++ct)
#pragma unroll
    for (int r = 0; r < 8; ++r) {
      int pix = wq * 16 + r + 8 * hi;
      int c = a * 128 + ct * 16 + ln;
      xT[pix * LDA_X + c] = f2bf(Oacc[ct][r] / rowl[r]);
    }
  if (PRE) {
    for (int id = tid; id < 1024; id += 256) {
      int pix = id >> 4, c8 = id & 15;
      copy16(gmc + (size_t)(m0 + pix) * CHAN + c8 * 8, xT + pix * LDA_X + 256 + c8 * 8);
    }
  } else {
    for (int idx = tid; idx < CHAN * 64; idx += 256) {
      int c = idx >> 6, pix = idx & 63;
      xT[pix * LDA_X + 256 + c] = f2bf(curb[(size_t)c * MPIX + m0 + pix]);
    }
  }
  if (tid < 64) red[tid] = 0.f;
  wait_async();
  __syncthreads();

  // ---- head: y = w1*x + b1 -> BN -> leaky -> w2 dot ----
  if (PRE) {
    // WMMA head: wave computes y slice [16 o x 64 pix], K = 384 (A + B pipelined)
    v8f yac[4];
#pragma unroll
    for (int pt = 0; pt < 4; ++pt) yac[pt] = zero;
    Frag16 af[12];
    {
      const unsigned short* wrow0 = w1bf + (size_t)(wave * 16 + ln) * 384 + hi * 8;
      af[0].q[0] = *(const v4i*)wrow0;
      af[0].q[1] = *(const v4i*)(wrow0 + 16);
#pragma unroll
      for (int kc = 0; kc < 12; ++kc) {
        if (kc < 11) {
          const unsigned short* wrow = wrow0 + (kc + 1) * 32;
          af[kc + 1].q[0] = *(const v4i*)wrow;
          af[kc + 1].q[1] = *(const v4i*)(wrow + 16);
        }
        Frag16 xb[4];
        const int xbase0 = ln * LDA_X + kc * 32 + hi * 8;
        xb[0].q[0] = *(const v4i*)&xT[xbase0];
        xb[0].q[1] = *(const v4i*)&xT[xbase0 + 16];
#pragma unroll
        for (int pt = 0; pt < 4; ++pt) {
          if (pt < 3) {
            int bbase = xbase0 + (pt + 1) * 16 * LDA_X;
            xb[pt + 1].q[0] = *(const v4i*)&xT[bbase];
            xb[pt + 1].q[1] = *(const v4i*)&xT[bbase + 16];
          }
          yac[pt] = __builtin_amdgcn_wmma_f32_16x16x32_bf16(false, af[kc].bf, false, xb[pt].bf,
                                                            (short)0, yac[pt], false, false);
        }
      }
    }
    float inv8[8], sh8[8], w28[8], b18[8];
#pragma unroll
    for (int r = 0; r < 8; ++r) {
      int o = wave * 16 + hi * 8 + r;
      float inv = gamma[o] * rsqrtf(rvar[o] + 1e-5f);
      inv8[r] = inv;
      sh8[r] = beta[o] - rmean[o] * inv;
      w28[r] = w2[o];
      b18[r] = b1[o];
    }
#pragma unroll
    for (int pt = 0; pt < 4; ++pt) {
      float part = 0.f;
#pragma unroll
      for (int r = 0; r < 8; ++r) {
        float y = (yac[pt][r] + b18[r]) * inv8[r] + sh8[r];
        y = y > 0.f ? y : 0.01f * y;
        part += w28[r] * y;
      }
      atomicAdd(&red[pt * 16 + ln], part);   // ds_add_f32
    }
    __syncthreads();
    if (tid < 64) out[(size_t)b * MPIX + m0 + tid] = red[tid] + b2[0];
  } else {
    // scalar fallback head
    float* redb = red;                       // 256 floats available
    int pix = tid >> 2, part = tid & 3;
    float acc = 0.f;
    for (int o = part * 32; o < part * 32 + 32; ++o) {
      const float* wrow = w1 + o * 384;
      float s = b1[o];
#pragma unroll 4
      for (int c = 0; c < 384; ++c) s += wrow[c] * bf2f(xT[pix * LDA_X + c]);
      float inv = gamma[o] * rsqrtf(rvar[o] + 1e-5f);
      float y = s * inv + (beta[o] - rmean[o] * inv);
      y = y > 0.f ? y : 0.01f * y;
      acc += w2[o] * y;
    }
    redb[tid] = acc;
    __syncthreads();
    if (part == 0)
      out[(size_t)b * MPIX + m0 + pix] =
          redb[tid] + redb[tid + 1] + redb[tid + 2] + redb[tid + 3] + b2[0];
  }
}

// ---- pre-pass: f32 -> bf16, same layout ----
__global__ void convert_cm(const float* __restrict__ s, unsigned short* __restrict__ d, int n) {
  for (int i = blockIdx.x * blockDim.x + threadIdx.x; i < n; i += gridDim.x * blockDim.x)
    d[i] = f2bf(s[i]);
}

// ---- pre-pass: f32 [b][c][m] -> bf16 [b][m][c] (32x32 LDS tile transpose) ----
__global__ __launch_bounds__(256)
void convert_mc(const float* __restrict__ ref, const float* __restrict__ cur,
                unsigned short* __restrict__ refMC, unsigned short* __restrict__ curMC) {
  __shared__ unsigned short t[32][33];
  int tx = threadIdx.x & 31, ty = threadIdx.x >> 5;
  int bz = blockIdx.z, b = bz >> 1;
  const float* src = ((bz & 1) ? cur : ref) + (size_t)b * CHAN * MPIX;
  unsigned short* dst = ((bz & 1) ? curMC : refMC) + (size_t)b * CHAN * MPIX;
  int m0 = blockIdx.x * 32, c0 = blockIdx.y * 32;
#pragma unroll
  for (int j = 0; j < 4; ++j) {
    int c = c0 + ty + j * 8;
    t[ty + j * 8][tx] = f2bf(src[(size_t)c * MPIX + m0 + tx]);
  }
  __syncthreads();
#pragma unroll
  for (int j = 0; j < 4; ++j) {
    int m = m0 + ty + j * 8;
    dst[(size_t)m * CHAN + c0 + tx] = t[tx][ty + j * 8];
  }
}

extern "C" void kernel_launch(void* const* d_in, const int* in_sizes, int n_in,
                              void* d_out, int out_size, void* d_ws, size_t ws_size,
                              hipStream_t stream) {
  (void)in_sizes; (void)n_in; (void)out_size;
  const float* ref  = (const float*)d_in[0];
  const float* cur  = (const float*)d_in[1];
  const float* w1   = (const float*)d_in[2];
  const float* b1   = (const float*)d_in[3];
  const float* gam  = (const float*)d_in[4];
  const float* bet  = (const float*)d_in[5];
  const float* rme  = (const float*)d_in[6];
  const float* rva  = (const float*)d_in[7];
  const float* w2   = (const float*)d_in[8];
  const float* b2   = (const float*)d_in[9];
  float* out = (float*)d_out;

  const size_t NELEM = (size_t)BATCH * CHAN * MPIX;          // 4,194,304
  const size_t NEED  = 3 * NELEM * 2 + (size_t)CHAN * 384 * 2;
  dim3 grid(MPIX / 64, BATCH);

  if (ws_size >= NEED) {
    unsigned short* curCM = (unsigned short*)d_ws;
    unsigned short* curMC = curCM + NELEM;
    unsigned short* refMC = curMC + NELEM;
    unsigned short* w1bf  = refMC + NELEM;
    convert_cm<<<2048, 256, 0, stream>>>(cur, curCM, (int)NELEM);
    convert_cm<<<96, 256, 0, stream>>>(w1, w1bf, CHAN * 384);
    convert_mc<<<dim3(MPIX / 32, CHAN / 32, BATCH * 2), 256, 0, stream>>>(ref, cur, refMC, curMC);
    anchor_attn_head<1><<<grid, 256, SHM_BYTES, stream>>>(ref, cur, w1, b1, gam, bet, rme, rva,
                                                          w2, b2, out, curCM, curMC, refMC, w1bf);
  } else {
    anchor_attn_head<0><<<grid, 256, SHM_BYTES, stream>>>(ref, cur, w1, b1, gam, bet, rme, rva,
                                                          w2, b2, out, nullptr, nullptr, nullptr,
                                                          nullptr);
  }
}
